// ChirpNet3_79465484910715
// MI455X (gfx1250) — matrix operand order repaired and checked by reading, not compiled, not run on hardware
//
#include <hip/hip_runtime.h>
#include <hip/hip_bf16.h>
#include <math.h>

typedef _Float16 v8h  __attribute__((ext_vector_type(8)));
typedef _Float16 v16h __attribute__((ext_vector_type(16)));
typedef float    v8f  __attribute__((ext_vector_type(8)));

#define D_INF 512
#define CHN   32
#define SEQL  256
#define DIMM  256
#define DI    512
#define NST   32
#define DTRK  16
#define CCH   16
#define BAT   8

enum { A_PLAIN = 0, A_CHIRP = 1, A_POSADD = 2, A_CONCAT = 3 };
enum { W_PLAIN = 0, W_CHIRP = 1 };
enum { ACT_NONE = 0, ACT_RELU = 1, ACT_SOFTPLUS = 2 };

// ---------------------------------------------------------------------------
// WMMA GEMM: C[M,N] = act(A[M,K] @ W[N,K]^T + bias)
// Block: 256 threads (8 wave32), C-tile 64x64; each wave owns 16x32 of C
// (one A fragment, two B fragments, two v_wmma_f32_16x16x32_f16 per K-slab).
// LDS tiles are stored pre-swizzled in WMMA fragment order so each lane
// fetches its fragment as two ds_load_b128; staging writes ds_store_b128.
// For lin0, the K axis is re-permuted to c-major (K' = c*512+f) so the
// 134 MB input streams with coalesced float4 loads and the scatter lands on
// the small L2-resident weight instead (WMODE = W_CHIRP).
// ---------------------------------------------------------------------------
template<int AMODE, int WMODE, int ACT>
__global__ __launch_bounds__(256)
void gemm_wmma(const float* __restrict__ A0, const float* __restrict__ A1,
               const float* __restrict__ W,  const float* __restrict__ bias,
               float* __restrict__ C, int M, int N, int K, int lda, int Ksplit)
{
    // fragment-order LDS: [subtile][lane][element j]
    __shared__ _Float16 Af[4][32][16];   // 4 M-subtiles of 16x32
    __shared__ _Float16 Bf[4][32][16];   // 4 N-subtiles of 16(cols)x32(K)

    const int tid  = threadIdx.x;
    const int m0   = blockIdx.y * 64;
    const int n0   = blockIdx.x * 64;
    const int lane = tid & 31;
    const int wv   = tid >> 5;
    const int wm   = wv & 3;             // wave's M-subtile (0..3)
    const int wn   = wv >> 2;            // wave's N-pair (0..1)

    // --- A staging coords: 64 rows x 4 k-octets, one octet per thread ---
    const int ar     = tid >> 2;                      // row in tile (0..63)
    const int ao     = tid & 3;                       // k-octet (0..3)
    const int a_tile = ar >> 4;
    const int a_lane = (ar & 15) + ((ao & 1) << 4);   // hi-half from octet
    const int a_j    = (ao >> 1) << 3;                // j base (0 or 8)
    // --- B staging coords: 4 n-subtiles x 32 lanes x 2 octets ---
    const int bo     = tid & 1;
    const int blane  = (tid >> 1) & 31;
    const int bt     = tid >> 6;                      // n-subtile (0..3)
    const int b_n    = (bt << 4) + (blane & 15);
    const int b_koff = ((blane >> 4) << 3) + (bo << 4);

    v8f acc0 = {}, acc1 = {};

    const int ksteps = (K + 31) >> 5;
    for (int ks = 0; ks < ksteps; ++ks) {
        const int k0 = ks << 5;

        // ---------------- stage A octet (8 contiguous K') ----------------
        {
            const int gm  = m0 + ar;
            const int gkb = k0 + (ao << 3);
            float va[8];
            if (gkb < K) {           // K is always a multiple of 8
                if (AMODE == A_CHIRP) {
                    // K' = c*512 + f ; A[m][k'] = x[b, c, l, f] (contiguous in f)
                    const int b = gm >> 8, l = gm & (SEQL - 1);
                    const int c = gkb >> 9, f = gkb & (D_INF - 1);
                    const float4* p = (const float4*)
                        (A0 + (size_t)(((b * CHN + c) * SEQL + l) * D_INF + f));
                    float4 x0 = p[0], x1 = p[1];
                    va[0]=x0.x; va[1]=x0.y; va[2]=x0.z; va[3]=x0.w;
                    va[4]=x1.x; va[5]=x1.y; va[6]=x1.z; va[7]=x1.w;
                } else if (AMODE == A_POSADD) {
                    const float4* p = (const float4*)(A0 + (size_t)gm * lda + gkb);
                    const float4* q = (const float4*)
                        (A1 + (size_t)(gm & (SEQL - 1)) * lda + gkb);
                    float4 x0 = p[0], x1 = p[1], y0 = q[0], y1 = q[1];
                    va[0]=x0.x+y0.x; va[1]=x0.y+y0.y; va[2]=x0.z+y0.z; va[3]=x0.w+y0.w;
                    va[4]=x1.x+y1.x; va[5]=x1.y+y1.y; va[6]=x1.z+y1.z; va[7]=x1.w+y1.w;
                } else { // A_PLAIN / A_CONCAT (octet never crosses Ksplit: mult of 8)
                    const float* base;
                    if (AMODE == A_CONCAT)
                        base = (gkb < Ksplit) ? (A0 + (size_t)gm * Ksplit + gkb)
                                              : (A1 + (size_t)gm * Ksplit + (gkb - Ksplit));
                    else
                        base = A0 + (size_t)gm * lda + gkb;
                    const float4* p = (const float4*)base;
                    float4 x0 = p[0], x1 = p[1];
                    va[0]=x0.x; va[1]=x0.y; va[2]=x0.z; va[3]=x0.w;
                    va[4]=x1.x; va[5]=x1.y; va[6]=x1.z; va[7]=x1.w;
                }
            } else {
                #pragma unroll
                for (int j = 0; j < 8; ++j) va[j] = 0.0f;
            }
            v8h h;
            #pragma unroll
            for (int j = 0; j < 8; ++j) h[j] = (_Float16)va[j];
            *(v8h*)&Af[a_tile][a_lane][a_j] = h;      // one ds_store_b128
        }
        // ---------------- stage B octet ----------------
        {
            const int gn = n0 + b_n;
            const int gk = k0 + b_koff;
            float vb[8];
            if (gn < N && gk < K) {
                if (WMODE == W_CHIRP) {
                    // original k = f*32 + c with K' = c*512 + f
                    const int c = gk >> 9, f = gk & (D_INF - 1);
                    const float* p = W + (size_t)gn * K + (f << 5) + c;
                    #pragma unroll
                    for (int j = 0; j < 8; ++j) vb[j] = p[j << 5];
                } else {
                    const float4* p = (const float4*)(W + (size_t)gn * K + gk);
                    float4 x0 = p[0], x1 = p[1];
                    vb[0]=x0.x; vb[1]=x0.y; vb[2]=x0.z; vb[3]=x0.w;
                    vb[4]=x1.x; vb[5]=x1.y; vb[6]=x1.z; vb[7]=x1.w;
                }
            } else {
                #pragma unroll
                for (int j = 0; j < 8; ++j) vb[j] = 0.0f;
            }
            v8h h;
            #pragma unroll
            for (int j = 0; j < 8; ++j) h[j] = (_Float16)vb[j];
            *(v8h*)&Bf[bt][blane][bo << 3] = h;       // one ds_store_b128
        }
        __syncthreads();

        // ---- fragments: contiguous 32B per lane -> 2x ds_load_b128 each ----
        v16h af  = *(const v16h*)&Af[wm][lane][0];
        v16h bf0 = *(const v16h*)&Bf[wn * 2][lane][0];
        v16h bf1 = *(const v16h*)&Bf[wn * 2 + 1][lane][0];
        acc0 = __builtin_amdgcn_wmma_f32_16x16x32_f16(false, af, false, bf0,
                                                      (short)0, acc0, false, false);
        acc1 = __builtin_amdgcn_wmma_f32_16x16x32_f16(false, af, false, bf1,
                                                      (short)0, acc1, false, false);
        __syncthreads();
    }

    // ---- epilogue: C layout = lane 0-15 -> M=r, lane 16-31 -> M=r+8 ----
    const int rl = lane & 15, hi = lane >> 4;
    const int mbase = m0 + wm * 16 + (hi << 3);
    #pragma unroll
    for (int t = 0; t < 2; ++t) {
        const int n = n0 + wn * 32 + t * 16 + rl;
        if (n < N) {
            const float bv = bias ? bias[n] : 0.0f;
            const v8f acc = t ? acc1 : acc0;
            #pragma unroll
            for (int r = 0; r < 8; ++r) {
                float v = acc[r] + bv;
                if (ACT == ACT_RELU)          v = fmaxf(v, 0.0f);
                else if (ACT == ACT_SOFTPLUS) v = (v > 20.0f) ? v : log1pf(__expf(v));
                C[(size_t)(mbase + r) * N + n] = v;
            }
        }
    }
}

// ---------------------------------------------------------------------------
// Row layernorm over DIM=256 features; one 256-thread block per row.
// ---------------------------------------------------------------------------
__global__ __launch_bounds__(256)
void layernorm_kernel(const float* __restrict__ in, const float* __restrict__ g,
                      const float* __restrict__ b, float* __restrict__ out)
{
    __shared__ float red[256];
    const int row = blockIdx.x, tid = threadIdx.x;
    float x = in[row * DIMM + tid];
    red[tid] = x; __syncthreads();
    for (int s = 128; s > 0; s >>= 1) { if (tid < s) red[tid] += red[tid + s]; __syncthreads(); }
    float mu = red[0] * (1.0f / DIMM); __syncthreads();
    float d = x - mu;
    red[tid] = d * d; __syncthreads();
    for (int s = 128; s > 0; s >>= 1) { if (tid < s) red[tid] += red[tid + s]; __syncthreads(); }
    float var = red[0] * (1.0f / DIMM);
    out[row * DIMM + tid] = d * rsqrtf(var + 1e-5f) * g[tid] + b[tid];
}

// ---------------------------------------------------------------------------
// Causal depthwise conv1d (k=4, left-pad 3) + SiLU over xz[:, :, :DI].
// ---------------------------------------------------------------------------
__global__ __launch_bounds__(256)
void dwconv_silu_kernel(const float* __restrict__ xz, const float* __restrict__ w,
                        const float* __restrict__ bias, float* __restrict__ xc)
{
    int t = blockIdx.x * 256 + threadIdx.x;   // 2048 * 512
    int d = t & (DI - 1);
    int m = t >> 9;
    int b = m >> 8, l = m & (SEQL - 1);
    float s = bias[d];
    #pragma unroll
    for (int j = 0; j < 4; ++j) {
        int ll = l - 3 + j;
        if (ll >= 0) s += xz[((b << 8) + ll) * (2 * DI) + d] * w[d * 4 + j];
    }
    xc[t] = s / (1.0f + __expf(-s));          // SiLU
}

// ---------------------------------------------------------------------------
// Selective scan: one wave32 per (batch, channel); lane = state index n.
// h = exp(dt*A)*h + dt*B*u ; y = (sum_n h*C + u*D) * silu(z).
// ---------------------------------------------------------------------------
__global__ __launch_bounds__(256)
void scan_kernel(const float* __restrict__ xc,   const float* __restrict__ dtb,
                 const float* __restrict__ Alog, const float* __restrict__ xdbl,
                 const float* __restrict__ Dv,   const float* __restrict__ xz,
                 float* __restrict__ y)
{
    const int wv   = blockIdx.x * 8 + (threadIdx.x >> 5);  // 0..4095 = b*DI+d
    const int lane = threadIdx.x & 31;                     // state n
    const int b = wv >> 9;
    const int d = wv & (DI - 1);
    const float A  = -__expf(Alog[d * NST + lane]);
    const float Dd = Dv[d];
    float h = 0.0f;
    for (int l = 0; l < SEQL; ++l) {
        const int m = (b << 8) + l;
        float dt = dtb[m * DI + d];
        float u  = xc[m * DI + d];
        float Bv = xdbl[m * 80 + DTRK + lane];
        float Cv = xdbl[m * 80 + DTRK + NST + lane];
        h = __expf(dt * A) * h + dt * Bv * u;
        float p = h * Cv;
        #pragma unroll
        for (int off = 16; off > 0; off >>= 1) p += __shfl_down(p, off, 32);
        if (lane == 0) {
            float zv = xz[m * (2 * DI) + DI + d];
            y[m * DI + d] = (p + u * Dd) * (zv / (1.0f + __expf(-zv)));
        }
    }
}

// ---------------------------------------------------------------------------
// Direct 3x3 SAME conv + ReLU, weights staged in LDS (<=9KB).
// ---------------------------------------------------------------------------
template<int IC>
__global__ __launch_bounds__(256)
void conv3x3_relu_kernel(const float* __restrict__ in, const float* __restrict__ w,
                         const float* __restrict__ bias, float* __restrict__ out)
{
    __shared__ float sw[CCH * IC * 9];
    __shared__ float sb[CCH];
    const int tid = threadIdx.x;
    for (int i = tid; i < CCH * IC * 9; i += 256) sw[i] = w[i];
    if (tid < CCH) sb[tid] = bias[tid];
    __syncthreads();

    const int gid = blockIdx.x * 256 + tid;   // 8*16*256*256
    const int x  = gid & 255;
    const int y  = (gid >> 8) & 255;
    const int oc = (gid >> 16) & 15;
    const int b  = gid >> 20;
    float s = sb[oc];
    for (int ic = 0; ic < IC; ++ic) {
        const float* ip = in + ((size_t)(b * IC + ic) * 256) * 256;
        const float* wp = sw + (oc * IC + ic) * 9;
        #pragma unroll
        for (int ky = 0; ky < 3; ++ky) {
            int yy = y + ky - 1;
            if ((unsigned)yy >= 256u) continue;
            #pragma unroll
            for (int kx = 0; kx < 3; ++kx) {
                int xx = x + kx - 1;
                if ((unsigned)xx >= 256u) continue;
                s += ip[yy * 256 + xx] * wp[ky * 3 + kx];
            }
        }
    }
    out[gid] = fmaxf(s, 0.0f);
}

// bilinear 256->128 == exact 2x2 mean pool
__global__ __launch_bounds__(256)
void pool2_kernel(const float* __restrict__ in, float* __restrict__ out)
{
    const int gid = blockIdx.x * 256 + threadIdx.x;  // 8*16*128*128
    const int x = gid & 127, y = (gid >> 7) & 127;
    const int c = (gid >> 14) & 15, b = gid >> 18;
    const float* ip = in + ((size_t)(b * CCH + c) * 256 + 2 * y) * 256 + 2 * x;
    out[gid] = 0.25f * (ip[0] + ip[1] + ip[256] + ip[257]);
}

// final 1x1 conv 16->1 (no relu)
__global__ __launch_bounds__(256)
void conv1x1_kernel(const float* __restrict__ in, const float* __restrict__ w,
                    const float* __restrict__ bias, float* __restrict__ out)
{
    const int gid = blockIdx.x * 256 + threadIdx.x;  // 8*128*128
    const int pix = gid & 16383;
    const int b   = gid >> 14;
    float s = bias[0];
    #pragma unroll
    for (int c = 0; c < CCH; ++c)
        s += in[(size_t)(b * CCH + c) * 16384 + pix] * w[c];
    out[gid] = s;
}

// ---------------------------------------------------------------------------
extern "C" void kernel_launch(void* const* d_in, const int* in_sizes, int n_in,
                              void* d_out, int out_size, void* d_ws, size_t ws_size,
                              hipStream_t stream)
{
    (void)in_sizes; (void)n_in; (void)out_size; (void)ws_size;
    const float* x         = (const float*)d_in[0];
    const float* lin0_w    = (const float*)d_in[1];
    const float* lin0_b    = (const float*)d_in[2];
    const float* ln0_g     = (const float*)d_in[3];
    const float* ln0_b     = (const float*)d_in[4];
    const float* lin1_w    = (const float*)d_in[5];
    const float* lin1_b    = (const float*)d_in[6];
    const float* ln1_g     = (const float*)d_in[7];
    const float* ln1_b     = (const float*)d_in[8];
    const float* pos       = (const float*)d_in[9];
    const float* m_in_w    = (const float*)d_in[10];
    const float* m_conv_w  = (const float*)d_in[11];
    const float* m_conv_b  = (const float*)d_in[12];
    const float* m_xproj_w = (const float*)d_in[13];
    const float* m_dt_w    = (const float*)d_in[14];
    const float* m_dt_b    = (const float*)d_in[15];
    const float* m_Alog    = (const float*)d_in[16];
    const float* m_D       = (const float*)d_in[17];
    const float* m_out_w   = (const float*)d_in[18];
    const float* blk_out_w = (const float*)d_in[19];
    const float* blk_out_b = (const float*)d_in[20];
    const float* dec_w     = (const float*)d_in[21];
    const float* dec_b     = (const float*)d_in[22];
    const float* c1_w      = (const float*)d_in[23];
    const float* c1_b      = (const float*)d_in[24];
    const float* c2_w      = (const float*)d_in[25];
    const float* c2_b      = (const float*)d_in[26];
    const float* c3_w      = (const float*)d_in[27];
    const float* c3_b      = (const float*)d_in[28];
    const float* c4_w      = (const float*)d_in[29];
    const float* c4_b      = (const float*)d_in[30];

    float* ws = (float*)d_ws;
    // [B*L = 2048 rows everywhere]
    float* t0   = ws;             // 2048*256   (also xd in decode stage)
    float* t1   = ws +  524288;   // 2048*256
    float* skip = ws + 1048576;   // 2048*256
    float* xzb  = ws + 1572864;   // 2048*1024
    float* xcb  = ws + 3670016;   // 2048*512
    float* xdbl = ws + 4718592;   // 2048*80
    float* dtb  = ws + 4882432;   // 2048*512
    float* ybuf = ws + 5931008;   // 2048*512
    float* xs2  = ws + 6979584;   // 2048*256
    float* xs3  = ws + 7503872;   // 2048*256
    // conv stage (everything above except t0 is dead by then)
    float* cb1  = ws +  524288;   // 8*16*256*256
    float* cb2  = ws + 8912896;   // 8*16*256*256
    float* pl   = ws + 17301504;  // 8*16*128*128

    const dim3 blk(256);
    const int MBLK = 2048 / 64;   // grid.y for all GEMMs (M=2048)

    // 1) encode lin0: relu(permute(x) @ lin0_w^T + b)   M=2048 N=256 K=16384
    //    (K re-permuted c-major: coalesced streaming of x, scatter on weight)
    gemm_wmma<A_CHIRP, W_CHIRP, ACT_RELU><<<dim3(DIMM / 64, MBLK), blk, 0, stream>>>(
        x, nullptr, lin0_w, lin0_b, t0, 2048, DIMM, D_INF * CHN, D_INF * CHN, 0);
    // 2) layernorm 0
    layernorm_kernel<<<2048, 256, 0, stream>>>(t0, ln0_g, ln0_b, t1);
    // 3) lin1 + relu
    gemm_wmma<A_PLAIN, W_PLAIN, ACT_RELU><<<dim3(DIMM / 64, MBLK), blk, 0, stream>>>(
        t1, nullptr, lin1_w, lin1_b, t0, 2048, DIMM, DIMM, DIMM, 0);
    // 4) layernorm 1 -> skip
    layernorm_kernel<<<2048, 256, 0, stream>>>(t0, ln1_g, ln1_b, skip);
    // 5) xz = (skip + pos) @ m_in_w^T    N=1024 K=256 (pos-add fused)
    gemm_wmma<A_POSADD, W_PLAIN, ACT_NONE><<<dim3((2 * DI) / 64, MBLK), blk, 0, stream>>>(
        skip, pos, m_in_w, nullptr, xzb, 2048, 2 * DI, DIMM, DIMM, 0);
    // 6) causal depthwise conv + silu -> xc
    dwconv_silu_kernel<<<(2048 * DI) / 256, 256, 0, stream>>>(xzb, m_conv_w, m_conv_b, xcb);
    // 7) x_dbl = xc @ m_xproj_w^T       N=80 K=512
    gemm_wmma<A_PLAIN, W_PLAIN, ACT_NONE><<<dim3((80 + 63) / 64, MBLK), blk, 0, stream>>>(
        xcb, nullptr, m_xproj_w, nullptr, xdbl, 2048, 80, DI, DI, 0);
    // 8) dt = softplus(x_dbl[:, :16] @ m_dt_w^T + dt_b)   N=512 K=16 (lda=80)
    gemm_wmma<A_PLAIN, W_PLAIN, ACT_SOFTPLUS><<<dim3(DI / 64, MBLK), blk, 0, stream>>>(
        xdbl, nullptr, m_dt_w, m_dt_b, dtb, 2048, DI, DTRK, 80, 0);
    // 9) selective scan (+ u*D and silu(z) gating fused)
    scan_kernel<<<(BAT * DI) / 8, 256, 0, stream>>>(xcb, dtb, m_Alog, xdbl, m_D, xzb, ybuf);
    // 10) y @ m_out_w^T                 N=256 K=512
    gemm_wmma<A_PLAIN, W_PLAIN, ACT_NONE><<<dim3(DIMM / 64, MBLK), blk, 0, stream>>>(
        ybuf, nullptr, m_out_w, nullptr, xs2, 2048, DIMM, DI, DI, 0);
    // 11) blk_out
    gemm_wmma<A_PLAIN, W_PLAIN, ACT_NONE><<<dim3(DIMM / 64, MBLK), blk, 0, stream>>>(
        xs2, nullptr, blk_out_w, blk_out_b, xs3, 2048, DIMM, DIMM, DIMM, 0);
    // 12) xd = relu(concat(xs3, skip) @ dec_w^T + b)   K=512 (concat fused)
    gemm_wmma<A_CONCAT, W_PLAIN, ACT_RELU><<<dim3(DIMM / 64, MBLK), blk, 0, stream>>>(
        xs3, skip, dec_w, dec_b, t0, 2048, DIMM, 2 * DIMM, 2 * DIMM, DIMM);
    // 13-15) conv stack on [8,*,256,256]
    conv3x3_relu_kernel<1 ><<<(BAT * CCH * 256 * 256) / 256, 256, 0, stream>>>(t0,  c1_w, c1_b, cb1);
    conv3x3_relu_kernel<16><<<(BAT * CCH * 256 * 256) / 256, 256, 0, stream>>>(cb1, c2_w, c2_b, cb2);
    conv3x3_relu_kernel<16><<<(BAT * CCH * 256 * 256) / 256, 256, 0, stream>>>(cb2, c3_w, c3_b, cb1);
    // 16) 2x2 mean pool -> [8,16,128,128]
    pool2_kernel<<<(BAT * CCH * 128 * 128) / 256, 256, 0, stream>>>(cb1, pl);
    // 17) 1x1 conv -> d_out [8,1,128,128]
    conv1x1_kernel<<<(BAT * 128 * 128) / 256, 256, 0, stream>>>(pl, c4_w, c4_b, (float*)d_out);
}
